// DreamerActorCritic_73572789781066
// MI455X (gfx1250) — compile-verified, matrix operand order checked by reading, not run on hardware
//
#include <hip/hip_runtime.h>
#include <cstddef>

// ---------------------------------------------------------------------------
// Types for WMMA
// ---------------------------------------------------------------------------
typedef __attribute__((ext_vector_type(16))) __bf16 bf16x16;
typedef __attribute__((ext_vector_type(8)))  __bf16 bf16x8;
typedef __attribute__((ext_vector_type(8)))  float  f32x8;

union FragBF { bf16x16 v; bf16x8 h[2]; };

#define LAMW 0.95f
#define ACTENT 3e-4f
#define DAC_T 334
#define DAC_B 64
#define DAC_TM1 333
#define DAC_N (DAC_B * DAC_TM1)   // 21312
#define DAC_M (DAC_B * DAC_T)     // 21376

// ---------------------------------------------------------------------------
// f32 -> bf16 convert
// ---------------------------------------------------------------------------
__global__ void dac_cvt(const float* __restrict__ in, __bf16* __restrict__ out, long n) {
  long i = (long)blockIdx.x * blockDim.x + threadIdx.x;
  if (i < n) out[i] = (__bf16)in[i];
}

// f32 (K x Nreal, row-major) -> bf16 transposed (Npad x K), zero-pad rows n>=Nreal
__global__ void dac_cvt_t(const float* __restrict__ W, __bf16* __restrict__ Wt,
                          int K, int Nreal, int Npad) {
  long i = (long)blockIdx.x * blockDim.x + threadIdx.x;
  long tot = (long)Npad * K;
  if (i >= tot) return;
  int n = (int)(i / K);
  int k = (int)(i % K);
  float v = (n < Nreal) ? W[(size_t)k * Nreal + n] : 0.0f;
  Wt[i] = (__bf16)v;
}

// ---------------------------------------------------------------------------
// bf16 WMMA GEMM: Out(M x N, f32, ld=ldOut) = X(M x 1024 bf16) @ Wt^T
// Wt is stored N x 1024 (column-major of the logical 1024 x N weight).
// Block tile 128(M) x (32*NFRAG)(N); 8 waves as 4(m) x 2(n);
// wave tile 32 x (16*NFRAG): 2 M-frags x NFRAG N-frags.
// Grid: (N/(32*NFRAG), M/128). Optional bias (length Nreal; cols>=Nreal get 0).
// ---------------------------------------------------------------------------
template <int NFRAG>
__global__ __launch_bounds__(256)
void dac_gemm_bf16(const __bf16* __restrict__ X, const __bf16* __restrict__ Wt,
                   float* __restrict__ Out, const float* __restrict__ bias,
                   int ldOut, int Nreal) {
  const int lane = threadIdx.x & 31;
  const int wid  = threadIdx.x >> 5;
  const int lr   = lane & 15;
  const int hi   = lane >> 4;           // 0 or 1
  const int kSel = hi * 8;              // lane's K sub-offset within frag
  const int mBase = blockIdx.y * 128 + (wid >> 1) * 32;
  const int nBase = blockIdx.x * (32 * NFRAG) + (wid & 1) * (16 * NFRAG);

  f32x8 c[2][NFRAG];
  #pragma unroll
  for (int i = 0; i < 2; ++i)
    #pragma unroll
    for (int j = 0; j < NFRAG; ++j)
      c[i][j] = (f32x8){0.f,0.f,0.f,0.f,0.f,0.f,0.f,0.f};

  const __bf16* xrow0 = X  + (size_t)(mBase + lr) * 1024;
  const __bf16* wrow0 = Wt + (size_t)(nBase + lr) * 1024;

  for (int kk = 0; kk < 1024; kk += 32) {
    FragBF a[2], b[NFRAG];
    #pragma unroll
    for (int i = 0; i < 2; ++i) {
      const __bf16* p = xrow0 + (size_t)i * 16 * 1024 + kk + kSel;
      a[i].h[0] = *(const bf16x8*)(p);
      a[i].h[1] = *(const bf16x8*)(p + 16);
    }
    #pragma unroll
    for (int j = 0; j < NFRAG; ++j) {
      const __bf16* p = wrow0 + (size_t)j * 16 * 1024 + kk + kSel;
      b[j].h[0] = *(const bf16x8*)(p);
      b[j].h[1] = *(const bf16x8*)(p + 16);
    }
    #pragma unroll
    for (int i = 0; i < 2; ++i)
      #pragma unroll
      for (int j = 0; j < NFRAG; ++j)
        c[i][j] = __builtin_amdgcn_wmma_f32_16x16x32_bf16(
            false, a[i].v, false, b[j].v, (short)0, c[i][j], false, false);
  }

  #pragma unroll
  for (int i = 0; i < 2; ++i)
    #pragma unroll
    for (int j = 0; j < NFRAG; ++j) {
      int col = nBase + j * 16 + lr;
      float badd = 0.0f;
      if (bias != nullptr) badd = (col < Nreal) ? bias[col] : 0.0f;
      #pragma unroll
      for (int r = 0; r < 8; ++r) {
        int row = mBase + i * 16 + hi * 8 + r;
        Out[(size_t)row * ldOut + col] = c[i][j][r] + badd;
      }
    }
}

// ---------------------------------------------------------------------------
// Row-wise rmsnorm (over 1024) + SiLU, f32 in -> bf16 out. Grid = M rows.
// ---------------------------------------------------------------------------
__global__ __launch_bounds__(256)
void dac_rms_silu(const float* __restrict__ H, const float* __restrict__ g,
                  __bf16* __restrict__ out) {
  const int row = blockIdx.x;
  const int tid = threadIdx.x;
  const float* x = H + (size_t)row * 1024;
  float v[4]; float ss = 0.0f;
  #pragma unroll
  for (int cidx = 0; cidx < 4; ++cidx) { v[cidx] = x[tid + cidx * 256]; ss += v[cidx] * v[cidx]; }
  __shared__ float red[256];
  red[tid] = ss;
  for (int s = 128; s > 0; s >>= 1) { __syncthreads(); if (tid < s) red[tid] += red[tid + s]; }
  __syncthreads();
  float rs = rsqrtf(red[0] * (1.0f / 1024.0f) + 1e-6f);
  #pragma unroll
  for (int cidx = 0; cidx < 4; ++cidx) {
    int n = tid + cidx * 256;
    float y = v[cidx] * rs * g[n];
    float sil = y / (1.0f + expf(-y));
    out[(size_t)row * 1024 + n] = (__bf16)sil;
  }
}

// ---------------------------------------------------------------------------
// Twohot bins / helpers
// ---------------------------------------------------------------------------
__device__ __forceinline__ float dac_symlog(float y) { return copysignf(log1pf(fabsf(y)), y); }
__device__ __forceinline__ float dac_symexp(float v) { return copysignf(expm1f(fabsf(v)), v); }
__device__ __forceinline__ float dac_bin(int i) {
  return dac_symexp(-20.0f + 40.0f * (float)i / 254.0f);
}

// twohot NLL given row logits l (255 valid, log-partition logZ)
__device__ float dac_twohot_nll(const float* __restrict__ l, float logZ, float y) {
  float blim = expm1f(20.0f);
  y = fminf(fmaxf(y, -blim), blim);
  float p = (dac_symlog(y) + 20.0f) * (254.0f / 40.0f);
  int k = (int)floorf(p);
  k = max(0, min(k, 253));
  float lo = dac_bin(k), hiv = dac_bin(k + 1);
  float w = fminf(fmaxf((y - lo) / (hiv - lo), 0.0f), 1.0f);
  return -((1.0f - w) * (l[k] - logZ) + w * (l[k + 1] - logZ));
}

// ---------------------------------------------------------------------------
// twohot_mean over slow logits (rows x 256 f32, 255 valid). Grid = M rows.
// ---------------------------------------------------------------------------
__global__ __launch_bounds__(256)
void dac_thmean(const float* __restrict__ logits, float* __restrict__ smean) {
  const int row = blockIdx.x;
  const int tid = threadIdx.x;
  const float* l = logits + (size_t)row * 256;
  __shared__ float red[256];
  __shared__ float sM, sZ;
  float lv = (tid < 255) ? l[tid] : -3.4e38f;
  red[tid] = lv;
  for (int s = 128; s > 0; s >>= 1) { __syncthreads(); if (tid < s) red[tid] = fmaxf(red[tid], red[tid + s]); }
  __syncthreads();
  if (tid == 0) sM = red[0];
  __syncthreads();
  float e = (tid < 255) ? expf(lv - sM) : 0.0f;
  red[tid] = e;
  for (int s = 128; s > 0; s >>= 1) { __syncthreads(); if (tid < s) red[tid] += red[tid + s]; }
  __syncthreads();
  if (tid == 0) sZ = red[0];
  __syncthreads();
  red[tid] = (tid < 255) ? e * dac_bin(tid) : 0.0f;
  for (int s = 128; s > 0; s >>= 1) { __syncthreads(); if (tid < s) red[tid] += red[tid + s]; }
  __syncthreads();
  if (tid == 0) smean[row] = red[0] / sZ;
}

// ---------------------------------------------------------------------------
// Serial per-batch-row scan: cumprod weight and reverse lambda-return.
// Launch <<<1, 64>>>.
// ---------------------------------------------------------------------------
__global__ void dac_scan(const float* __restrict__ reward, const float* __restrict__ cont,
                         const float* __restrict__ smean,
                         float* __restrict__ ret, float* __restrict__ wgt) {
  int b = threadIdx.x;
  if (b >= DAC_B) return;
  float prod = 1.0f;
  for (int t = 0; t < DAC_TM1; ++t) {
    prod *= cont[b * DAC_T + t];
    wgt[b * DAC_TM1 + t] = prod;
  }
  float r = smean[b * DAC_T + (DAC_T - 1)];
  for (int t = DAC_T - 2; t >= 0; --t) {
    float live = cont[b * DAC_T + t + 1];
    float interm = reward[b * DAC_T + t + 1] + (1.0f - LAMW) * live * smean[b * DAC_T + t + 1];
    r = interm + live * LAMW * r;
    ret[b * DAC_TM1 + t] = r;
  }
}

// ---------------------------------------------------------------------------
// Percentile (5th & 95th, linear interp) over ret[DAC_N] via binary search
// on order-preserving uint32 keys. Single block of 1024 threads.
// ---------------------------------------------------------------------------
__device__ __forceinline__ unsigned dac_okey(float f) {
  unsigned u = __float_as_uint(f);
  return (u & 0x80000000u) ? ~u : (u | 0x80000000u);
}
__device__ __forceinline__ float dac_inv_okey(unsigned k) {
  unsigned u = (k & 0x80000000u) ? (k ^ 0x80000000u) : ~k;
  return __uint_as_float(u);
}

__global__ __launch_bounds__(1024)
void dac_percentile(const float* __restrict__ ret, float* __restrict__ rsc) {
  const int n = DAC_N;
  const int tid = threadIdx.x;
  __shared__ int cnt[1024];
  __shared__ unsigned sLo, sHi;
  __shared__ float vals[4];
  double pos0 = 0.05 * (double)(n - 1);
  double pos1 = 0.95 * (double)(n - 1);
  long i0 = (long)pos0, i1 = (long)pos1;
  long ranks[4] = { i0, i0 + 1, i1, i1 + 1 };

  for (int r = 0; r < 4; ++r) {
    if (tid == 0) { sLo = 0u; sHi = 0xFFFFFFFFu; }
    for (int it = 0; it < 33; ++it) {
      __syncthreads();
      unsigned lo = sLo, hiu = sHi;
      unsigned mid = lo + ((hiu - lo) >> 1);
      int c = 0;
      for (int i = tid; i < n; i += 1024) c += (dac_okey(ret[i]) <= mid) ? 1 : 0;
      cnt[tid] = c;
      for (int s = 512; s > 0; s >>= 1) { __syncthreads(); if (tid < s) cnt[tid] += cnt[tid + s]; }
      __syncthreads();
      if (tid == 0 && lo < hiu) {
        if (cnt[0] >= (int)ranks[r] + 1) sHi = mid; else sLo = mid + 1;
      }
    }
    __syncthreads();
    if (tid == 0) vals[r] = dac_inv_okey(sLo);
    __syncthreads();
  }
  if (tid == 0) {
    float f0 = (float)(pos0 - (double)i0);
    float f1 = (float)(pos1 - (double)i1);
    float lo_p = vals[0] + f0 * (vals[1] - vals[0]);
    float hi_p = vals[2] + f1 * (vals[3] - vals[2]);
    rsc[0] = fmaxf(hi_p - lo_p, 1.0f);
  }
}

// ---------------------------------------------------------------------------
// Policy loss per (b,t), t < T-1. Grid = DAC_N blocks x 64 threads.
// ---------------------------------------------------------------------------
__global__ __launch_bounds__(64)
void dac_ploss(const float* __restrict__ plog, const int* __restrict__ act,
               const float* __restrict__ ret, const float* __restrict__ smean,
               const float* __restrict__ wgt, const float* __restrict__ rsc,
               float* __restrict__ plos) {
  const int i = blockIdx.x;
  const int tid = threadIdx.x;
  const int b = i / DAC_TM1, t = i % DAC_TM1;
  const int row = b * DAC_T + t;
  const float* l = plog + (size_t)row * 64;
  __shared__ float red[64];
  __shared__ float sLogp[64];
  __shared__ float sM, sZ;
  float lv = l[tid];
  red[tid] = lv;
  for (int s = 32; s > 0; s >>= 1) { __syncthreads(); if (tid < s) red[tid] = fmaxf(red[tid], red[tid + s]); }
  __syncthreads();
  if (tid == 0) sM = red[0];
  __syncthreads();
  float e = expf(lv - sM);
  red[tid] = e;
  for (int s = 32; s > 0; s >>= 1) { __syncthreads(); if (tid < s) red[tid] += red[tid + s]; }
  __syncthreads();
  if (tid == 0) sZ = red[0];
  __syncthreads();
  float p = 0.99f * (e / sZ) + 0.01f / 64.0f;
  float lp = logf(p);
  sLogp[tid] = lp;
  red[tid] = -p * lp;
  for (int s = 32; s > 0; s >>= 1) { __syncthreads(); if (tid < s) red[tid] += red[tid + s]; }
  __syncthreads();
  if (tid == 0) {
    float ent = red[0];
    int a = act[row];
    float adv = (ret[i] - smean[row]) / rsc[0];
    plos[i] = wgt[i] * (-(sLogp[a] * adv + ACTENT * ent));
  }
}

// ---------------------------------------------------------------------------
// Value loss per (b,t), t < T-1. Grid = DAC_N blocks x 256 threads.
// ---------------------------------------------------------------------------
__global__ __launch_bounds__(256)
void dac_vloss(const float* __restrict__ vlog, const float* __restrict__ ret,
               const float* __restrict__ smean, const float* __restrict__ wgt,
               float* __restrict__ vlos) {
  const int i = blockIdx.x;
  const int tid = threadIdx.x;
  const int b = i / DAC_TM1, t = i % DAC_TM1;
  const int row = b * DAC_T + t;
  const float* l = vlog + (size_t)row * 256;
  __shared__ float red[256];
  __shared__ float sM, sZ;
  float lv = (tid < 255) ? l[tid] : -3.4e38f;
  red[tid] = lv;
  for (int s = 128; s > 0; s >>= 1) { __syncthreads(); if (tid < s) red[tid] = fmaxf(red[tid], red[tid + s]); }
  __syncthreads();
  if (tid == 0) sM = red[0];
  __syncthreads();
  float e = (tid < 255) ? expf(lv - sM) : 0.0f;
  red[tid] = e;
  for (int s = 128; s > 0; s >>= 1) { __syncthreads(); if (tid < s) red[tid] += red[tid + s]; }
  __syncthreads();
  if (tid == 0) sZ = red[0];
  __syncthreads();
  if (tid == 0) {
    float logZ = sM + logf(sZ);
    float v = dac_twohot_nll(l, logZ, ret[i]) + dac_twohot_nll(l, logZ, smean[row]);
    vlos[i] = wgt[i] * v;
  }
}

// ---------------------------------------------------------------------------
// Final deterministic reduction: out[0] = (sum plos + sum vlos) / DAC_N
// ---------------------------------------------------------------------------
__global__ __launch_bounds__(1024)
void dac_final(const float* __restrict__ plos, const float* __restrict__ vlos,
               float* __restrict__ out) {
  const int tid = threadIdx.x;
  __shared__ float red[1024];
  float s = 0.0f;
  for (int i = tid; i < DAC_N; i += 1024) s += plos[i] + vlos[i];
  red[tid] = s;
  for (int st = 512; st > 0; st >>= 1) { __syncthreads(); if (tid < st) red[tid] += red[tid + st]; }
  __syncthreads();
  if (tid == 0) out[0] = red[0] / (float)DAC_N;
}

// ---------------------------------------------------------------------------
// Host launcher
// ---------------------------------------------------------------------------
extern "C" void kernel_launch(void* const* d_in, const int* in_sizes, int n_in,
                              void* d_out, int out_size, void* d_ws, size_t ws_size,
                              hipStream_t stream) {
  (void)in_sizes; (void)n_in; (void)out_size; (void)ws_size;
  const int M = DAC_M;                    // 21376 = 167 * 128
  const size_t MK = (size_t)M * 1024;

  const float* feat   = (const float*)d_in[0];
  const float* reward = (const float*)d_in[1];
  const float* cont   = (const float*)d_in[2];
  const int*   act    = (const int*)d_in[3];
  const float* pW_in  = (const float*)d_in[4];
  const float* pW_hid = (const float*)d_in[5];
  const float* p_gn   = (const float*)d_in[6];
  const float* pHw    = (const float*)d_in[7];
  const float* pHb    = (const float*)d_in[8];
  const float* vW_in  = (const float*)d_in[9];
  const float* vW_hid = (const float*)d_in[10];
  const float* v_gn   = (const float*)d_in[11];
  const float* vHw    = (const float*)d_in[12];
  const float* vHb    = (const float*)d_in[13];
  const float* sW_in  = (const float*)d_in[14];
  const float* sW_hid = (const float*)d_in[15];
  const float* s_gn   = (const float*)d_in[16];
  const float* sHw    = (const float*)d_in[17];
  const float* sHb    = (const float*)d_in[18];
  float* out = (float*)d_out;

  char* ws = (char*)d_ws;
  size_t off = 0;
  auto wsalloc = [&](size_t bytes) -> char* {
    char* p = ws + off;
    off = (off + bytes + 255) & ~(size_t)255;
    return p;
  };
  __bf16* featB = (__bf16*)wsalloc(MK * 2);
  __bf16* actA  = (__bf16*)wsalloc(MK * 2);
  __bf16* actB  = (__bf16*)wsalloc(MK * 2);
  float*  Hbuf  = (float*) wsalloc(MK * 4);
  __bf16* Wt    = (__bf16*)wsalloc((size_t)1024 * 1024 * 2);
  __bf16* hWt   = (__bf16*)wsalloc((size_t)256 * 1024 * 2);
  float*  plog  = (float*) wsalloc((size_t)M * 64 * 4);
  float*  vlog  = (float*) wsalloc((size_t)M * 256 * 4);   // shared: slow then value
  float*  smean = (float*) wsalloc((size_t)M * 4);
  float*  retb  = (float*) wsalloc((size_t)DAC_N * 4);
  float*  wgtb  = (float*) wsalloc((size_t)DAC_N * 4);
  float*  rsc   = (float*) wsalloc(256);
  float*  plos  = (float*) wsalloc((size_t)DAC_N * 4);
  float*  vlos  = (float*) wsalloc((size_t)DAC_N * 4);

  // feat -> bf16
  dac_cvt<<<(int)((MK + 255) / 256), 256, 0, stream>>>(feat, featB, (long)MK);

  // Run a 5-layer MLP (F=U=1024) returning pointer to last bf16 activation.
  auto run_mlp = [&](const float* Win, const float* Whid, const float* gn) -> const __bf16* {
    const __bf16* cur = featB;
    __bf16* pp[2] = { actA, actB };
    for (int l = 0; l < 5; ++l) {
      const float* W = (l == 0) ? Win : (Whid + (size_t)(l - 1) * 1024 * 1024);
      dac_cvt_t<<<4096, 256, 0, stream>>>(W, Wt, 1024, 1024, 1024);
      dac_gemm_bf16<4><<<dim3(8, 167), 256, 0, stream>>>(cur, Wt, Hbuf, nullptr, 1024, 1024);
      dac_rms_silu<<<M, 256, 0, stream>>>(Hbuf, gn + (size_t)l * 1024, pp[l & 1]);
      cur = pp[l & 1];
    }
    return cur;
  };

  // Slow MLP + head -> slow logits (in vlog buffer) -> slow_mean
  {
    const __bf16* hs = run_mlp(sW_in, sW_hid, s_gn);
    dac_cvt_t<<<1024, 256, 0, stream>>>(sHw, hWt, 1024, 255, 256);
    dac_gemm_bf16<4><<<dim3(2, 167), 256, 0, stream>>>(hs, hWt, vlog, sHb, 256, 255);
    dac_thmean<<<M, 256, 0, stream>>>(vlog, smean);
  }

  // Value MLP + head -> value logits (reuse vlog buffer; slow logits no longer needed)
  {
    const __bf16* hv = run_mlp(vW_in, vW_hid, v_gn);
    dac_cvt_t<<<1024, 256, 0, stream>>>(vHw, hWt, 1024, 255, 256);
    dac_gemm_bf16<4><<<dim3(2, 167), 256, 0, stream>>>(hv, hWt, vlog, vHb, 256, 255);
  }

  // Policy MLP + head -> policy logits (N=64 -> narrow 2-frag variant)
  {
    const __bf16* hp = run_mlp(pW_in, pW_hid, p_gn);
    dac_cvt_t<<<256, 256, 0, stream>>>(pHw, hWt, 1024, 64, 64);
    dac_gemm_bf16<2><<<dim3(1, 167), 256, 0, stream>>>(hp, hWt, plog, pHb, 64, 64);
  }

  // Lambda-return + cumprod weight (serial per batch row)
  dac_scan<<<1, 64, 0, stream>>>(reward, cont, smean, retb, wgtb);

  // Return scale from 5th/95th percentiles
  dac_percentile<<<1, 1024, 0, stream>>>(retb, rsc);

  // Losses
  dac_ploss<<<DAC_N, 64, 0, stream>>>(plog, act, retb, smean, wgtb, rsc, plos);
  dac_vloss<<<DAC_N, 256, 0, stream>>>(vlog, retb, smean, wgtb, vlos);

  // Final scalar
  dac_final<<<1, 1024, 0, stream>>>(plos, vlos, out);
}
